// WITT_ADALN_Encoder_3865470566409
// MI455X (gfx1250) — matrix operand order repaired and checked by reading, not compiled, not run on hardware
//
#include <hip/hip_runtime.h>

// ---------------- model constants ----------------
#define B_   16
#define C_   256
#define NH_  8
#define WS_  8
#define SS_  4
#define N_   64
#define HD_  32
#define TOK  65536          // B * H * W  (= B * NWIN * N)

typedef _Float16 v16h __attribute__((ext_vector_type(16)));
typedef _Float16 h8   __attribute__((ext_vector_type(8)));
typedef float    v8f  __attribute__((ext_vector_type(8)));

union Frag { v16h v; h8 h[2]; };

// ---------------- small prep kernels ----------------

// W (K x N, f32 row-major) -> Wt (N x K, f16) so each WMMA B-fragment lane
// reads two runs of 8 contiguous f16 along K (two b128 loads).
__global__ void cvt_transpose(const float* __restrict__ W, _Float16* __restrict__ Wt,
                              int K, int N) {
  int idx = blockIdx.x * blockDim.x + threadIdx.x;
  if (idx >= K * N) return;
  int k = idx / N, n = idx - k * N;
  Wt[(size_t)n * K + k] = (_Float16)W[idx];
}

// ada = silu(cond) @ w_ada + b_ada   (16 x 1536, trivially small)
__global__ void ada_kernel(const float* __restrict__ cond, const float* __restrict__ wada,
                           const float* __restrict__ bada, float* __restrict__ ada) {
  int idx = blockIdx.x * blockDim.x + threadIdx.x;
  if (idx >= B_ * 1536) return;
  int b = idx / 1536, n = idx - b * 1536;
  float acc = bada[n];
  for (int k = 0; k < C_; ++k) {
    float c = cond[b * C_ + k];
    acc += (c / (1.f + __expf(-c))) * wada[k * 1536 + n];
  }
  ada[idx] = acc;
}

// ---------------- LN + AdaLN modulate (+ optional roll/window scatter) ----------------
// One wave per token (wave32: lane handles 8 contiguous channels), 8 tokens/block.
__global__ __launch_bounds__(256) void ln_mod_kernel(
    const float* __restrict__ src, const float* __restrict__ gamma,
    const float* __restrict__ beta, const float* __restrict__ ada,
    int shiftOff, int scaleOff, int windowed, _Float16* __restrict__ out) {
  int wave = threadIdx.x >> 5, lane = threadIdx.x & 31;
  int t = blockIdx.x * 8 + wave;
  int b = t >> 12;
  size_t srcBase;
  if (windowed) {
    int win = (t >> 6) & 63, n = t & 63;
    int hh = ((win >> 3) << 3) + (n >> 3);
    int ww = ((win & 7) << 3) + (n & 7);
    int sh = (hh + SS_) & 63, sw = (ww + SS_) & 63;     // rolled[p] = x[(p+SS)%64]
    srcBase = ((size_t)b * 4096 + sh * 64 + sw) * C_;
  } else {
    srcBase = (size_t)t * C_;
  }
  float vals[8];
  float s = 0.f, s2 = 0.f;
#pragma unroll
  for (int i = 0; i < 8; ++i) {
    float v = src[srcBase + lane * 8 + i];
    vals[i] = v; s += v; s2 += v * v;
  }
#pragma unroll
  for (int off = 1; off < 32; off <<= 1) {
    s  += __shfl_xor(s,  off, 32);
    s2 += __shfl_xor(s2, off, 32);
  }
  float mean = s * (1.f / 256.f);
  float var  = s2 * (1.f / 256.f) - mean * mean;
  float rstd = rsqrtf(var + 1e-5f);
#pragma unroll
  for (int i = 0; i < 8; ++i) {
    int c = lane * 8 + i;
    float h = (vals[i] - mean) * rstd * gamma[c] + beta[c];
    float sc = ada[b * 1536 + scaleOff + c];
    float sh = ada[b * 1536 + shiftOff + c];
    out[(size_t)t * C_ + c] = (_Float16)(h * (1.f + sc) + sh);
  }
}

// ---------------- WMMA GEMM: C(MxN,f32) = A(MxK,f16) @ Wt(NxK,f16)^T ----------------
// Each wave computes a 32x64 macro-tile (2 m-tiles x 4 n-tiles, 8 f32 accs):
// each A fragment feeds 4 WMMAs, each B fragment feeds 2 -> 12 b128 loads per
// 8 WMMAs. 8 waves/block cover 256 rows x 64 cols; B fragments are shared
// across waves via WGP$. Epilogue modes:
//   0: +bias -> f16            (QKV)
//   1: gelu(+bias) -> f16      (fc1)
//   2: window-reverse + roll(+SS) scatter, x + gm*(v+bias) -> f32  (proj)
//   3: xmid + gp*(v+bias) -> f32                                    (fc2/out)
__global__ __launch_bounds__(256) void gemm_wmma(
    const _Float16* __restrict__ A, const _Float16* __restrict__ Wt,
    const float* __restrict__ bias, _Float16* __restrict__ outH,
    float* __restrict__ outF, const float* __restrict__ resid,
    const float* __restrict__ gate, int N, int K, int mode) {
  int wave = threadIdx.x >> 5, lane = threadIdx.x & 31;
  int half = lane >> 4, lcol = lane & 15;
  int nbase = blockIdx.x * 64;
  int mbase = (blockIdx.y * 16 + wave * 2) * 16;   // rows [mbase, mbase+32)

  const v8f vzero = {0.f, 0.f, 0.f, 0.f, 0.f, 0.f, 0.f, 0.f};
  v8f acc[2][4];
#pragma unroll
  for (int i = 0; i < 2; ++i)
#pragma unroll
    for (int j = 0; j < 4; ++j) acc[i][j] = vzero;

  // A-fragment lane layout: lanes 0-15 row=lcol, K offsets {0..7,16..23};
  // lanes 16-31 row=lcol, K offsets {8..15,24..31}.
  const _Float16* arow0 = A + (size_t)(mbase + lcol) * K + half * 8;
  const _Float16* arow1 = arow0 + (size_t)16 * K;
  // B-fragment: lanes 0-15 col=lcol K {0..7,16..23}; lanes 16-31 K {8..15,24..31}.
  const _Float16* brow  = Wt + (size_t)(nbase + lcol) * K + half * 8;

  for (int kc = 0; kc < K; kc += 32) {
    Frag fa0, fa1;
    fa0.h[0] = *(const h8*)(arow0 + kc);
    fa0.h[1] = *(const h8*)(arow0 + kc + 16);
    fa1.h[0] = *(const h8*)(arow1 + kc);
    fa1.h[1] = *(const h8*)(arow1 + kc + 16);
#pragma unroll
    for (int j = 0; j < 4; ++j) {
      Frag fb;
      const _Float16* bp = brow + (size_t)j * 16 * K + kc;
      fb.h[0] = *(const h8*)(bp);
      fb.h[1] = *(const h8*)(bp + 16);
      acc[0][j] = __builtin_amdgcn_wmma_f32_16x16x32_f16(false, fa0.v, false, fb.v,
                                                         (short)0, acc[0][j], false, false);
      acc[1][j] = __builtin_amdgcn_wmma_f32_16x16x32_f16(false, fa1.v, false, fb.v,
                                                         (short)0, acc[1][j], false, false);
    }
  }

#pragma unroll
  for (int i = 0; i < 2; ++i) {
#pragma unroll
    for (int j = 0; j < 4; ++j) {
      int col = nbase + j * 16 + lcol;
      float bc = bias[col];
#pragma unroll
      for (int r = 0; r < 8; ++r) {
        int row = mbase + i * 16 + r + half * 8;  // C/D layout: lanes>=16 hold M=r+8
        float v = acc[i][j][r] + bc;
        if (mode == 0) {
          outH[(size_t)row * N + col] = (_Float16)v;
        } else if (mode == 1) {
          outH[(size_t)row * N + col] = (_Float16)(0.5f * v * (1.f + erff(v * 0.70710678f)));
        } else if (mode == 2) {
          int b = row >> 12, win = (row >> 6) & 63, n = row & 63;
          int hh = ((win >> 3) << 3) + (n >> 3);
          int ww = ((win & 7) << 3) + (n & 7);
          int dh = (hh + SS_) & 63, dw = (ww + SS_) & 63;   // roll(+SS)
          size_t dst = ((size_t)b * 4096 + dh * 64 + dw) * C_ + col;
          outF[dst] = resid[dst] + gate[b * 1536 + col] * v;
        } else {
          size_t dst = (size_t)row * C_ + col;
          outF[dst] = resid[dst] + gate[(row >> 12) * 1536 + col] * v;
        }
      }
    }
  }
}

// ---------------- windowed attention ----------------
__device__ __forceinline__ int regid(int p) { return p < 56 ? 0 : (p < 60 ? 1 : 2); }

// One block per (batch,window); wave = head. N=64 tokens, HD=32.
// V^T staged to LDS once (coalesced b128 loads) so V B-fragments are two
// contiguous ds_load_b128 each; P staged via LDS to convert C->A layout.
__global__ __launch_bounds__(256) void attn_wmma(
    const _Float16* __restrict__ qkv, const float* __restrict__ rel_bias,
    _Float16* __restrict__ outp) {
  __shared__ __align__(16) _Float16 vbuf[8][2048];   // per-head V^T: [d][token]
  __shared__ __align__(16) _Float16 pbuf[8][1024];   // per-head 16x64 P staging
  int wid = blockIdx.x;
  int winimg = wid & 63;
  int wh = winimg >> 3, ww = winimg & 7;
  int h = threadIdx.x >> 5, lane = threadIdx.x & 31;
  int half = lane >> 4, lcol = lane & 15;
  const float scale = 0.17677669529663689f;          // 32^-0.5

  // stage V^T: vbuf[h][d*64 + t] = V[t][d]
  {
    int t4 = lane >> 2;            // token sub-index 0..7
    int d0 = (lane & 3) * 8;       // 8 head-dims per lane
#pragma unroll
    for (int i = 0; i < 8; ++i) {
      int t = i * 8 + t4;
      h8 vv = *(const h8*)(qkv + (size_t)(wid * 64 + t) * 768 + 512 + h * 32 + d0);
#pragma unroll
      for (int j = 0; j < 8; ++j) vbuf[h][(d0 + j) * 64 + t] = vv[j];
    }
  }
  __syncthreads();

  for (int mi = 0; mi < 4; ++mi) {
    int mbase = mi * 16;
    Frag fq;
    {
      const _Float16* qrow =
          qkv + (size_t)(wid * 64 + mbase + lcol) * 768 + h * 32 + half * 8;
      fq.h[0] = *(const h8*)(qrow);
      fq.h[1] = *(const h8*)(qrow + 16);
    }
    v8f S[4];
#pragma unroll
    for (int nj = 0; nj < 4; ++nj) {
      const _Float16* krow =
          qkv + (size_t)(wid * 64 + nj * 16 + lcol) * 768 + 256 + h * 32 + half * 8;
      Frag fk;
      fk.h[0] = *(const h8*)(krow);
      fk.h[1] = *(const h8*)(krow + 16);
      v8f z = {0.f, 0.f, 0.f, 0.f, 0.f, 0.f, 0.f, 0.f};
      S[nj] = __builtin_amdgcn_wmma_f32_16x16x32_f16(false, fq.v, false, fk.v,
                                                     (short)0, z, false, false);
    }
    // rel-pos bias + shifted-window mask (module constants, recomputed analytically)
#pragma unroll
    for (int nj = 0; nj < 4; ++nj) {
#pragma unroll
      for (int r = 0; r < 8; ++r) {
        int qr = mbase + r + half * 8;
        int kcid = nj * 16 + lcol;
        int qy = qr >> 3, qx = qr & 7, ky = kcid >> 3, kx = kcid & 7;
        float rb = rel_bias[((qy - ky + 7) * 15 + (qx - kx + 7)) * NH_ + h];
        int ivq = regid(wh * 8 + qy) * 3 + regid(ww * 8 + qx);
        int ivk = regid(wh * 8 + ky) * 3 + regid(ww * 8 + kx);
        float mk = (ivq != ivk) ? -100.f : 0.f;
        S[nj][r] = S[nj][r] * scale + rb + mk;
      }
    }
    // row softmax: each 16-lane half holds one row per accumulator slot
#pragma unroll
    for (int r = 0; r < 8; ++r) {
      float mx = fmaxf(fmaxf(S[0][r], S[1][r]), fmaxf(S[2][r], S[3][r]));
#pragma unroll
      for (int off = 1; off < 16; off <<= 1) mx = fmaxf(mx, __shfl_xor(mx, off, 16));
      float sum = 0.f;
#pragma unroll
      for (int nj = 0; nj < 4; ++nj) { float e = __expf(S[nj][r] - mx); S[nj][r] = e; sum += e; }
#pragma unroll
      for (int off = 1; off < 16; off <<= 1) sum += __shfl_xor(sum, off, 16);
      float inv = 1.f / sum;
#pragma unroll
      for (int nj = 0; nj < 4; ++nj) S[nj][r] *= inv;
    }
    // stage P (C-layout -> row-major LDS) for reuse as A-fragment
#pragma unroll
    for (int nj = 0; nj < 4; ++nj)
#pragma unroll
      for (int r = 0; r < 8; ++r)
        pbuf[h][(r + half * 8) * 64 + nj * 16 + lcol] = (_Float16)S[nj][r];
    __syncthreads();
    // O(16x32) = P(16x64) @ V(64x32)
#pragma unroll
    for (int dj = 0; dj < 2; ++dj) {
      v8f O = {0.f, 0.f, 0.f, 0.f, 0.f, 0.f, 0.f, 0.f};
#pragma unroll
      for (int kc = 0; kc < 64; kc += 32) {
        Frag fp;
        const _Float16* prow = &pbuf[h][lcol * 64 + kc + half * 8];
        fp.h[0] = *(const h8*)(prow);
        fp.h[1] = *(const h8*)(prow + 16);
        Frag fv;
        const _Float16* vp = &vbuf[h][(dj * 16 + lcol) * 64 + kc + half * 8];
        fv.h[0] = *(const h8*)(vp);
        fv.h[1] = *(const h8*)(vp + 16);
        O = __builtin_amdgcn_wmma_f32_16x16x32_f16(false, fp.v, false, fv.v,
                                                   (short)0, O, false, false);
      }
#pragma unroll
      for (int r = 0; r < 8; ++r) {
        int row = mbase + r + half * 8;
        outp[(size_t)(wid * 64 + row) * C_ + h * 32 + dj * 16 + lcol] = (_Float16)O[r];
      }
    }
    __syncthreads();
  }
}

// ---------------- launcher ----------------
extern "C" void kernel_launch(void* const* d_in, const int* in_sizes, int n_in,
                              void* d_out, int out_size, void* d_ws, size_t ws_size,
                              hipStream_t stream) {
  (void)in_sizes; (void)n_in; (void)out_size; (void)ws_size;
  const float* x     = (const float*)d_in[0];
  const float* cond  = (const float*)d_in[1];
  const float* n1g   = (const float*)d_in[2];
  const float* n1b   = (const float*)d_in[3];
  const float* wqkv  = (const float*)d_in[4];
  const float* bqkv  = (const float*)d_in[5];
  const float* relb  = (const float*)d_in[6];
  const float* wproj = (const float*)d_in[7];
  const float* bproj = (const float*)d_in[8];
  const float* n2g   = (const float*)d_in[9];
  const float* n2b   = (const float*)d_in[10];
  const float* wfc1  = (const float*)d_in[11];
  const float* bfc1  = (const float*)d_in[12];
  const float* wfc2  = (const float*)d_in[13];
  const float* bfc2  = (const float*)d_in[14];
  const float* wada  = (const float*)d_in[15];
  const float* bada  = (const float*)d_in[16];

  char* ws = (char*)d_ws;
  size_t off = 0;
  float*    adab   = (float*)(ws + off);    off += (size_t)B_ * 1536 * 4;
  _Float16* wtqkv  = (_Float16*)(ws + off); off += (size_t)768 * 256 * 2;
  _Float16* wtproj = (_Float16*)(ws + off); off += (size_t)256 * 256 * 2;
  _Float16* wtfc1  = (_Float16*)(ws + off); off += (size_t)1024 * 256 * 2;
  _Float16* wtfc2  = (_Float16*)(ws + off); off += (size_t)256 * 1024 * 2;
  off = (off + 255) & ~(size_t)255;
  _Float16* hbuf   = (_Float16*)(ws + off); off += (size_t)TOK * 256 * 2;   // hwin / h2
  _Float16* qbuf   = (_Float16*)(ws + off); off += (size_t)TOK * 1024 * 2;  // qkv / mlp hidden
  _Float16* abuf   = (_Float16*)(ws + off); off += (size_t)TOK * 256 * 2;   // attn out
  float*    xmid   = (float*)(ws + off);    off += (size_t)TOK * 256 * 4;

  // weight prep + AdaLN modulation
  cvt_transpose<<<(256 * 768 + 255) / 256, 256, 0, stream>>>(wqkv, wtqkv, 256, 768);
  cvt_transpose<<<(256 * 256 + 255) / 256, 256, 0, stream>>>(wproj, wtproj, 256, 256);
  cvt_transpose<<<(256 * 1024 + 255) / 256, 256, 0, stream>>>(wfc1, wtfc1, 256, 1024);
  cvt_transpose<<<(1024 * 256 + 255) / 256, 256, 0, stream>>>(wfc2, wtfc2, 1024, 256);
  ada_kernel<<<(B_ * 1536 + 255) / 256, 256, 0, stream>>>(cond, wada, bada, adab);

  // LN1 + modulate + roll + window partition (f16)
  ln_mod_kernel<<<TOK / 8, 256, 0, stream>>>(x, n1g, n1b, adab, 0, 256, 1, hbuf);
  // QKV GEMM (65536 x 768 x 256)
  gemm_wmma<<<dim3(768 / 64, TOK / 256), 256, 0, stream>>>(
      hbuf, wtqkv, bqkv, qbuf, nullptr, nullptr, nullptr, 768, 256, 0);
  // attention per window
  attn_wmma<<<B_ * 64, 256, 0, stream>>>(qbuf, relb, abuf);
  // proj GEMM + window reverse + roll back + gm-gated residual -> xmid
  gemm_wmma<<<dim3(256 / 64, TOK / 256), 256, 0, stream>>>(
      abuf, wtproj, bproj, nullptr, xmid, x, adab + 512, 256, 256, 2);
  // LN2 + modulate (f16)
  ln_mod_kernel<<<TOK / 8, 256, 0, stream>>>(xmid, n2g, n2b, adab, 768, 1024, 0, hbuf);
  // fc1 GEMM + exact GELU (65536 x 1024 x 256)
  gemm_wmma<<<dim3(1024 / 64, TOK / 256), 256, 0, stream>>>(
      hbuf, wtfc1, bfc1, qbuf, nullptr, nullptr, nullptr, 1024, 256, 1);
  // fc2 GEMM + gp-gated residual -> out (65536 x 256 x 1024)
  gemm_wmma<<<dim3(256 / 64, TOK / 256), 256, 0, stream>>>(
      qbuf, wtfc2, bfc2, nullptr, (float*)d_out, xmid, adab + 1280, 256, 1024, 3);
}
